// MMoE_60112362275421
// MI455X (gfx1250) — matrix operand order repaired and checked by reading, not compiled, not run on hardware
//
#include <hip/hip_runtime.h>
#include <hip/hip_bf16.h>
#include <math.h>

// ---------------- problem constants ----------------
#define NTOK   6144      // 128 * 48 tokens
#define BATCH  128
#define SEQ    48
#define DDIM   768
#define HDIM   768
#define NEXP   8
#define NTASK  5
#define KTOP   4

// ---------------- WMMA types ----------------
typedef __attribute__((ext_vector_type(16))) __bf16 v16bf;
typedef __attribute__((ext_vector_type(8)))  float  v8f;

union Frag {
    v16bf v;
    unsigned int   u[8];
    unsigned short s[16];
};

__device__ __forceinline__ unsigned short f2bf(float f) {
    unsigned int u = __float_as_uint(f);
    unsigned int r = u + 0x7FFFu + ((u >> 16) & 1u);   // round-to-nearest-even
    return (unsigned short)(r >> 16);
}

// pack two floats into one dword of bf16s: [31:16]=hi, [15:0]=lo
__device__ __forceinline__ unsigned int pack2bf(float lo, float hi) {
    return (unsigned int)f2bf(lo) | ((unsigned int)f2bf(hi) << 16);
}

// ---------------- utility: zero fill ----------------
__global__ void zero_kernel(float* __restrict__ p, int n) {
    int i = blockIdx.x * blockDim.x + threadIdx.x;
    if (i < n) p[i] = 0.0f;
}

// ---------------- gating: logits -> top4 softmax -> dense gates [T,N,E] ----------------
__global__ __launch_bounds__(256)
void gate_kernel(const float* __restrict__ x, const float* __restrict__ gw,
                 const float* __restrict__ gb, float* __restrict__ gates) {
    int wave = threadIdx.x >> 5;
    int lane = threadIdx.x & 31;
    int tok  = blockIdx.x * 8 + wave;
    const float* xr = x + (size_t)tok * DDIM;

    for (int t = 0; t < NTASK; ++t) {
        float logit[NEXP];
#pragma unroll
        for (int e = 0; e < NEXP; ++e) {
            const float* w = gw + ((size_t)(t * NEXP + e)) * DDIM;
            float p = 0.f;
            for (int d = lane; d < DDIM; d += 32) p += xr[d] * w[d];
#pragma unroll
            for (int off = 16; off > 0; off >>= 1) p += __shfl_xor(p, off, 32);
            logit[e] = p + gb[t * NEXP + e];
        }
        // serial top-4 (all lanes redundantly; ties -> first index, matching top_k)
        int mask = 0; float tv[KTOP]; int ti[KTOP];
#pragma unroll
        for (int k = 0; k < KTOP; ++k) {
            float best = -3.4e38f; int bi = 0;
#pragma unroll
            for (int e = 0; e < NEXP; ++e)
                if (!((mask >> e) & 1) && logit[e] > best) { best = logit[e]; bi = e; }
            tv[k] = best; ti[k] = bi; mask |= 1 << bi;
        }
        float mx = tv[0], s = 0.f;
#pragma unroll
        for (int k = 0; k < KTOP; ++k) { tv[k] = expf(tv[k] - mx); s += tv[k]; }
        if (lane < NEXP) {
            float gv = 0.f;
#pragma unroll
            for (int k = 0; k < KTOP; ++k) if (ti[k] == lane) gv = tv[k] / s;
            gates[((size_t)t * NTOK + tok) * NEXP + lane] = gv;
        }
    }
}

// ---------------- cv^2 loss over importance + load ----------------
__device__ __forceinline__ float cv_squared(const float* v) {
    float m = 0.f;
#pragma unroll
    for (int e = 0; e < NEXP; ++e) m += v[e];
    m *= (1.0f / NEXP);
    float var = 0.f;
#pragma unroll
    for (int e = 0; e < NEXP; ++e) { float d = v[e] - m; var += d * d; }
    var *= (1.0f / (NEXP - 1));          // unbiased (ddof=1), matches torch .var()
    return var / (m * m + 1e-10f);
}

__global__ void loss_kernel(const float* __restrict__ gates, float* __restrict__ out) {
    __shared__ float imp[NTASK * NEXP];
    __shared__ float ldc[NTASK * NEXP];
    int p = threadIdx.x;
    if (p < NTASK * NEXP) {
        int t = p / NEXP, e = p % NEXP;
        float si = 0.f, sl = 0.f;
        for (int n = 0; n < NTOK; ++n) {
            float g = gates[((size_t)t * NTOK + n) * NEXP + e];
            si += g;
            if (g > 0.f) sl += 1.f;
        }
        imp[p] = si; ldc[p] = sl;
    }
    __syncthreads();
    if (p == 0) {
        float loss = 0.f;
        for (int t = 0; t < NTASK; ++t)
            loss += cv_squared(&imp[t * NEXP]) + cv_squared(&ldc[t * NEXP]);
        out[BATCH] = loss * 0.01f;       // mmoe_total_loss slot
    }
}

// ---------------- WMMA bf16 GEMM, fused expert epilogues ----------------
// MODE 0: out = relu(A@B + bias)                       (layer 1 -> h)
// MODE 1: out += gate[task][row][e] * (A@B + bias)     (layer 2 -> combine)
#define BM 128
#define BN 128
#define BK 32
#define LDA 40
#define LDB 40
#define TI 2      // M WMMA tiles per wave (32 rows)
#define TJ 4      // N WMMA tiles per wave (64 cols)

template <int MODE>
__global__ __launch_bounds__(256)
void gemm_expert(const float* __restrict__ A, const float* __restrict__ Bm,
                 const float* __restrict__ bias, float* __restrict__ out,
                 const float* __restrict__ gates, const int* __restrict__ task_p,
                 int expert, int Kdim, int Ndim) {
    __shared__ unsigned short As[BM * LDA];   // A tile, row-major [m][k], bf16
    __shared__ unsigned short Bs[BN * LDB];   // B tile, K-transposed [n][k], bf16

    const int tid  = threadIdx.x;
    const int lane = tid & 31;
    const int wave = tid >> 5;
    const int wm   = wave >> 1;        // 0..3 : 32-row strip
    const int wn   = wave & 1;         // 0..1 : 64-col strip
    const int kh   = lane >> 4;        // lane half for fragment layout
    const int l15  = lane & 15;
    const int bm   = blockIdx.y * BM;
    const int bn   = blockIdx.x * BN;
    const int KT   = Kdim / BK;

    v8f acc[TI][TJ];
#pragma unroll
    for (int i = 0; i < TI; ++i)
#pragma unroll
        for (int j = 0; j < TJ; ++j)
#pragma unroll
            for (int k = 0; k < 8; ++k) acc[i][j][k] = 0.f;

    for (int kt = 0; kt < KT; ++kt) {
        const int k0 = kt * BK;
        if (kt + 1 < KT) {
            __builtin_prefetch(A + (size_t)(bm + (tid >> 3)) * Kdim + k0 + BK + (tid & 7) * 4, 0, 3);
            __builtin_prefetch(Bm + (size_t)(k0 + BK + (tid >> 5) * 2) * Ndim + bn + (tid & 31) * 4, 0, 3);
        }
        // ---- stage A tile (128x32 f32 -> bf16), packed b32 LDS stores ----
        {
            const int col = (tid & 7) * 4;          // k offset, multiple of 4
            const int row = tid >> 3;               // 0..31
#pragma unroll
            for (int rr = 0; rr < 4; ++rr) {
                const int r = row + rr * 32;
                float4 v = *(const float4*)(A + (size_t)(bm + r) * Kdim + k0 + col);
                unsigned int* dst = (unsigned int*)&As[r * LDA + col];
                dst[0] = pack2bf(v.x, v.y);
                dst[1] = pack2bf(v.z, v.w);
            }
        }
        // ---- stage B tile (32x128 f32), K-transposed + K-pair-packed b32 stores ----
        {
            const int n4 = (tid & 31) * 4;          // column group (4 cols), 0..124
            const int kp = tid >> 5;                // 0..7 -> k pairs
#pragma unroll
            for (int rr = 0; rr < 2; ++rr) {
                const int k = (kp + rr * 8) * 2;    // even k
                float4 v0 = *(const float4*)(Bm + (size_t)(k0 + k)     * Ndim + bn + n4);
                float4 v1 = *(const float4*)(Bm + (size_t)(k0 + k + 1) * Ndim + bn + n4);
                *(unsigned int*)&Bs[(n4 + 0) * LDB + k] = pack2bf(v0.x, v1.x);
                *(unsigned int*)&Bs[(n4 + 1) * LDB + k] = pack2bf(v0.y, v1.y);
                *(unsigned int*)&Bs[(n4 + 2) * LDB + k] = pack2bf(v0.z, v1.z);
                *(unsigned int*)&Bs[(n4 + 3) * LDB + k] = pack2bf(v0.w, v1.w);
            }
        }
        __syncthreads();

        Frag af[TI], bf[TJ];
#pragma unroll
        for (int t = 0; t < TI; ++t) {
            const int row = wm * 32 + t * 16 + l15;
            // A 16x32 bf16 layout: halves hold K = 8*kh.. and 16+8*kh..
            *(uint4*)&af[t].u[0] = *(const uint4*)&As[row * LDA + 8 * kh];
            *(uint4*)&af[t].u[4] = *(const uint4*)&As[row * LDA + 16 + 8 * kh];
        }
#pragma unroll
        for (int t = 0; t < TJ; ++t) {
            const int nn = wn * 64 + t * 16 + l15;
            // B 32x16 bf16 layout: halves hold K = 8*kh.. and 16+8*kh..
            *(uint4*)&bf[t].u[0] = *(const uint4*)&Bs[nn * LDB + 8 * kh];
            *(uint4*)&bf[t].u[4] = *(const uint4*)&Bs[nn * LDB + 16 + 8 * kh];
        }
#pragma unroll
        for (int i = 0; i < TI; ++i)
#pragma unroll
            for (int j = 0; j < TJ; ++j)
                acc[i][j] = __builtin_amdgcn_wmma_f32_16x16x32_bf16(
                    false, af[i].v, false, bf[j].v,
                    (short)0, acc[i][j], false, false);
        __syncthreads();
    }

    // ---- epilogue ----
    const int t = (MODE == 1) ? *task_p : 0;
#pragma unroll
    for (int i = 0; i < TI; ++i) {
        const int row0 = bm + wm * 32 + i * 16;
        float g[8];
        if (MODE == 1) {
#pragma unroll
            for (int r = 0; r < 8; ++r)
                g[r] = gates[((size_t)t * NTOK + row0 + r + 8 * kh) * NEXP + expert];
        }
#pragma unroll
        for (int j = 0; j < TJ; ++j) {
            const int col = bn + wn * 64 + j * 16 + l15;
            const float bb = bias[col];
#pragma unroll
            for (int r = 0; r < 8; ++r) {
                const int row = row0 + r + 8 * kh;
                float v = acc[i][j][r] + bb;
                if (MODE == 0) {
                    v = v > 0.f ? v : 0.f;
                    out[(size_t)row * Ndim + col] = v;
                } else {
                    out[(size_t)row * Ndim + col] += v * g[r];
                }
            }
        }
    }
}

// ---------------- seq-sum + LayerNorm + head + sigmoid (block per batch row) ----------------
__global__ __launch_bounds__(256)
void finalize_kernel(const float* __restrict__ y, const float* __restrict__ ln_s,
                     const float* __restrict__ ln_b, const float* __restrict__ hw,
                     const float* __restrict__ hb, float* __restrict__ out) {
    const int b = blockIdx.x, tid = threadIdx.x;
    float v[3];
#pragma unroll
    for (int j = 0; j < 3; ++j) {
        const int d = tid + j * 256;
        const float* p = y + (size_t)b * SEQ * DDIM + d;
        float a = 0.f;
        for (int s = 0; s < SEQ; ++s) a += p[(size_t)s * DDIM];
        v[j] = a;
    }
    __shared__ float red[256];
    // mean
    red[tid] = v[0] + v[1] + v[2];
    __syncthreads();
    for (int o = 128; o > 0; o >>= 1) { if (tid < o) red[tid] += red[tid + o]; __syncthreads(); }
    const float mean = red[0] * (1.0f / DDIM);
    __syncthreads();
    // var (population, ddof=0)
    red[tid] = v[0] * v[0] + v[1] * v[1] + v[2] * v[2];
    __syncthreads();
    for (int o = 128; o > 0; o >>= 1) { if (tid < o) red[tid] += red[tid + o]; __syncthreads(); }
    const float var = red[0] * (1.0f / DDIM) - mean * mean;
    __syncthreads();
    const float rstd = rsqrtf(var + 1e-5f);
    // layernorm + head dot
    float part = 0.f;
#pragma unroll
    for (int j = 0; j < 3; ++j) {
        const int d = tid + j * 256;
        const float f = (v[j] - mean) * rstd * ln_s[d] + ln_b[d];
        part += f * hw[d];
    }
    red[tid] = part;
    __syncthreads();
    for (int o = 128; o > 0; o >>= 1) { if (tid < o) red[tid] += red[tid + o]; __syncthreads(); }
    if (tid == 0) {
        const float o = red[0] + hb[0];
        out[b] = 1.0f / (1.0f + expf(-o));
    }
}

// ---------------- MSE pred loss ----------------
__global__ void predloss_kernel(const float* __restrict__ true_y, float* __restrict__ out) {
    __shared__ float red[BATCH];
    const int t = threadIdx.x;
    const float d = out[t] - true_y[t];
    red[t] = d * d;
    __syncthreads();
    for (int o = BATCH / 2; o > 0; o >>= 1) { if (t < o) red[t] += red[t + o]; __syncthreads(); }
    if (t == 0) out[BATCH + 1] = red[0] * (1.0f / BATCH);
}

// ---------------- launcher ----------------
extern "C" void kernel_launch(void* const* d_in, const int* in_sizes, int n_in,
                              void* d_out, int out_size, void* d_ws, size_t ws_size,
                              hipStream_t stream) {
    (void)in_sizes; (void)n_in; (void)out_size; (void)ws_size;

    const float* x      = (const float*)d_in[0];   // [128,48,768] == [6144,768]
    const float* true_y = (const float*)d_in[1];   // [128,1]
    const float* gate_w = (const float*)d_in[2];   // [5,8,768]
    const float* gate_b = (const float*)d_in[3];   // [5,8]
    const float* W1     = (const float*)d_in[4];   // [8,768,768]
    const float* b1     = (const float*)d_in[5];   // [8,768]
    const float* W2     = (const float*)d_in[6];   // [8,768,768]
    const float* b2     = (const float*)d_in[7];   // [8,768]
    const float* ln_s   = (const float*)d_in[8];   // [768]
    const float* ln_b   = (const float*)d_in[9];   // [768]
    const float* head_w = (const float*)d_in[10];  // [768,1]
    const float* head_b = (const float*)d_in[11];  // [1]
    const int*   task   = (const int*)d_in[12];    // scalar
    float* out = (float*)d_out;                    // [130]: scores, moe_loss, pred_loss

    // workspace layout
    float* h_buf = (float*)d_ws;                         // [6144*768]
    float* ybuf  = h_buf + (size_t)NTOK * HDIM;          // [6144*768]
    float* gates = ybuf + (size_t)NTOK * DDIM;           // [5*6144*8]

    const int yN = NTOK * DDIM;
    zero_kernel<<<(yN + 255) / 256, 256, 0, stream>>>(ybuf, yN);

    gate_kernel<<<NTOK / 8, 256, 0, stream>>>(x, gate_w, gate_b, gates);
    loss_kernel<<<1, 64, 0, stream>>>(gates, out);

    dim3 gg(HDIM / BN, NTOK / BM);     // 6 x 48 blocks
    dim3 bb(256);
    for (int e = 0; e < NEXP; ++e) {
        gemm_expert<0><<<gg, bb, 0, stream>>>(
            x, W1 + (size_t)e * DDIM * HDIM, b1 + (size_t)e * HDIM,
            h_buf, nullptr, nullptr, e, DDIM, HDIM);
        gemm_expert<1><<<gg, bb, 0, stream>>>(
            h_buf, W2 + (size_t)e * HDIM * DDIM, b2 + (size_t)e * DDIM,
            ybuf, gates, task, e, HDIM, DDIM);
    }

    finalize_kernel<<<BATCH, 256, 0, stream>>>(ybuf, ln_s, ln_b, head_w, head_b, out);
    predloss_kernel<<<1, BATCH, 0, stream>>>(true_y, out);
}
